// MultiAggr_28252294873671
// MI455X (gfx1250) — matrix operand order repaired and checked by reading, not compiled, not run on hardware
//
#include <hip/hip_runtime.h>
#include <hip/hip_bf16.h>

typedef __attribute__((ext_vector_type(16))) _Float16 v16h;
typedef __attribute__((ext_vector_type(8)))  _Float16 v8h;
typedef __attribute__((ext_vector_type(4)))  _Float16 v4h;
typedef __attribute__((ext_vector_type(8)))  float    v8f;
typedef __attribute__((ext_vector_type(4)))  float    v4f;

#define NTOT   500000     // rows
#define HC     256        // hidden channels (K dim of GEMM1, out cols of GEMM2)
#define AC     64         // aggr channels
#define AK     320        // AC*5
#define NSEG   8192       // num graphs
#define RPB    128        // rows per block (tail: 500000 % 128 == 32)
#define NBLK   ((NTOT + RPB - 1) / RPB)
#define APAD   (HC + 8)   // LDS row stride in halves (528B: 16B aligned, bank-skewed)
#define GPB    8          // graphs per finalize block (W2 reuse factor)

// ---- order-preserving float<->uint encoding for atomic min/max ----
__device__ __forceinline__ unsigned encf(float x) {
    unsigned u = __float_as_uint(x);
    return (u & 0x80000000u) ? ~u : (u | 0x80000000u);
}
__device__ __forceinline__ float decf(unsigned u) {
    return __uint_as_float((u & 0x80000000u) ? (u & 0x7FFFFFFFu) : ~u);
}

// ---- kernel 1: W1 [HC, AK] f32 -> W1^T [AK, HC] f16 (column-major for B frags) ----
__global__ void k_transpose_w1(const float* __restrict__ W1, _Float16* __restrict__ Wt) {
    int n = blockIdx.x;      // 0..AK-1
    int k = threadIdx.x;     // 0..HC-1
    Wt[n * HC + k] = (_Float16)W1[k * AK + n];
}

// ---- kernel 2: init accumulators ----
__global__ void k_init(float* s, unsigned* mn, unsigned* mx, float* me,
                       float* m1, float* m2, int* cnt) {
    int i = blockIdx.x * blockDim.x + threadIdx.x;
    if (i < NSEG * AC) {
        s[i]  = 0.0f;
        mn[i] = 0xFFFFFFFFu;   // >= every encoding (acts as +inf)
        mx[i] = 0x00000000u;   // <= every encoding (acts as -inf)
        me[i] = 0.0f;
        m1[i] = 0.0f;
        m2[i] = 0.0f;
    }
    if (i < NSEG) cnt[i] = 0;
}

// ---- segmented flush of one C-tile half (8 rows held by this lane) ----
// segs sorted ascending within block, -1 padding only as suffix, so
// endpoint equality  <=>  all 8 rows in one segment (branch-free fast path).
__device__ __forceinline__ void aggr_flush(
        const v8f& c, const int* __restrict__ segs, int segbase, int aggr, int f,
        float* __restrict__ o_sum, unsigned* __restrict__ o_min,
        unsigned* __restrict__ o_max, float* __restrict__ o_mean,
        float* __restrict__ o_m1, float* __restrict__ o_m2) {
    const int s0 = segs[segbase];
    const int s7 = segs[segbase + 7];
    if (s0 == s7) {                       // ~89% of 8-row groups
        if (s0 < 0) return;               // fully padded tail group
        const int o = s0 * AC + f;
        if (aggr == 0 || aggr == 3) {
            float a = 0.0f;
#pragma unroll
            for (int v = 0; v < 8; ++v) a += c[v];
            atomicAdd(((aggr == 0) ? o_sum : o_mean) + o, a);
        } else if (aggr == 1) {
            float a = c[0];
#pragma unroll
            for (int v = 1; v < 8; ++v) a = fminf(a, c[v]);
            atomicMin(o_min + o, encf(a));
        } else if (aggr == 2) {
            float a = c[0];
#pragma unroll
            for (int v = 1; v < 8; ++v) a = fmaxf(a, c[v]);
            atomicMax(o_max + o, encf(a));
        } else {
            float a1 = 0.0f, a2 = 0.0f;
#pragma unroll
            for (int v = 0; v < 8; ++v) { a1 += c[v]; a2 += c[v] * c[v]; }
            atomicAdd(o_m1 + o, a1);
            atomicAdd(o_m2 + o, a2);
        }
    } else {                              // group straddles a segment boundary
#pragma unroll
        for (int v = 0; v < 8; ++v) {
            const int s = segs[segbase + v];
            if (s < 0) continue;
            const int o = s * AC + f;
            if      (aggr == 0) atomicAdd(o_sum + o, c[v]);
            else if (aggr == 1) atomicMin(o_min + o, encf(c[v]));
            else if (aggr == 2) atomicMax(o_max + o, encf(c[v]));
            else if (aggr == 3) atomicAdd(o_mean + o, c[v]);
            else { atomicAdd(o_m1 + o, c[v]); atomicAdd(o_m2 + o, c[v] * c[v]); }
        }
    }
}

// ---- kernel 3: fused GEMM1 (WMMA f16->f32) + segmented multi-aggregation ----
// 128 rows/block; 40 jobs = 20 col-tiles x 2 row-tile-quads; each job computes
// four 16x16 C tiles sharing one B fragment (4 WMMAs per B load pair).
// All 5 fragments are materialized before the WMMAs so the scheduler can
// clause the 10 loads per K-step and issue the 4 WMMAs back-to-back.
__launch_bounds__(256, 2)
__global__ void k_gemm_aggr(const float* __restrict__ x, const int* __restrict__ batch,
                            const _Float16* __restrict__ Wt,
                            float* __restrict__ o_sum, unsigned* __restrict__ o_min,
                            unsigned* __restrict__ o_max, float* __restrict__ o_mean,
                            float* __restrict__ o_m1, float* __restrict__ o_m2,
                            int* __restrict__ cnt) {
    __shared__ _Float16 As[RPB][APAD];
    __shared__ int segs[RPB];

    const int row0 = blockIdx.x * RPB;
    const int tid  = threadIdx.x;

    // stage segment ids (-1 pads the tail block)
    if (tid < RPB) {
        const int row = row0 + tid;
        segs[tid] = (row < NTOT) ? batch[row] : -1;
    }
    // stage 128x256 x-block, f32 -> f16, float4 loads / 8B LDS stores
    {
        const int c4 = (tid & 63) << 2;   // column group of 4
        const int r0 = tid >> 6;          // 0..3
#pragma unroll
        for (int i = 0; i < 32; ++i) {
            const int r   = r0 + (i << 2);
            const int row = row0 + r;
            v4f xv = {};
            if (row < NTOT) xv = *(const v4f*)&x[row * HC + c4];
            v4h hv;
#pragma unroll
            for (int e = 0; e < 4; ++e) hv[e] = (_Float16)xv[e];
            *(v4h*)&As[r][c4] = hv;
        }
    }
    __syncthreads();

    // run-based per-segment counting (~1-3 atomics per block, not per row)
    if (tid < RPB) {
        const int s = segs[tid];
        if (s >= 0 && (tid == 0 || segs[tid - 1] != s)) {
            int e = tid;
            while (e + 1 < RPB && segs[e + 1] == s) ++e;
            atomicAdd(&cnt[s], e - tid + 1);
        }
    }

    const int wave = tid >> 5;
    const int lane = tid & 31;
    const int col  = lane & 15;           // N index within 16x16 tile
    const int hi   = lane >> 4;           // half-wave select
    const int m0   = (wave & 1) * 64;     // base of this wave's 4 row-tiles

    const _Float16* const ar0 = &As[m0 +  0 + col][0];
    const _Float16* const ar1 = &As[m0 + 16 + col][0];
    const _Float16* const ar2 = &As[m0 + 32 + col][0];
    const _Float16* const ar3 = &As[m0 + 48 + col][0];

    auto load_a = [&](const _Float16* arow, int kk) -> v16h {
        const _Float16* pa = arow + kk + hi * 8;
        v8h a0 = *(const v8h*)(pa);
        v8h a1 = *(const v8h*)(pa + 16);
        v16h av;
#pragma unroll
        for (int e = 0; e < 8; ++e) { av[e] = a0[e]; av[e + 8] = a1[e]; }
        return av;
    };

    for (int ct = wave >> 1; ct < 20; ct += 4) {
        const _Float16* bcol = &Wt[(ct * 16 + col) * HC];
        v8f c0 = {}, c1 = {}, c2 = {}, c3 = {};
#pragma unroll
        for (int kk = 0; kk < HC; kk += 32) {
            // B fragment first (longest latency): lane = column ct*16+col,
            // k at kk+8*hi+[0..7] and kk+16+8*hi+[0..7], two 16B global reads.
            const _Float16* pb = bcol + kk + hi * 8;
            v8h b0 = *(const v8h*)(pb);
            v8h b1 = *(const v8h*)(pb + 16);
            v16h bv;
#pragma unroll
            for (int e = 0; e < 8; ++e) { bv[e] = b0[e]; bv[e + 8] = b1[e]; }

            // all four A fragments live simultaneously -> one load clause
            v16h a0 = load_a(ar0, kk);
            v16h a1 = load_a(ar1, kk);
            v16h a2 = load_a(ar2, kk);
            v16h a3 = load_a(ar3, kk);

            c0 = __builtin_amdgcn_wmma_f32_16x16x32_f16(false, a0, false, bv,
                                                        (short)0, c0, false, false);
            c1 = __builtin_amdgcn_wmma_f32_16x16x32_f16(false, a1, false, bv,
                                                        (short)0, c1, false, false);
            c2 = __builtin_amdgcn_wmma_f32_16x16x32_f16(false, a2, false, bv,
                                                        (short)0, c2, false, false);
            c3 = __builtin_amdgcn_wmma_f32_16x16x32_f16(false, a3, false, bv,
                                                        (short)0, c3, false, false);
        }

        const int aggr = ct >> 2;                 // 4 col-tiles per aggregator
        const int f    = ((ct & 3) << 4) + col;   // feature within aggregator
        aggr_flush(c0, segs, m0 +  0 + hi * 8, aggr, f, o_sum, o_min, o_max, o_mean, o_m1, o_m2);
        aggr_flush(c1, segs, m0 + 16 + hi * 8, aggr, f, o_sum, o_min, o_max, o_mean, o_m1, o_m2);
        aggr_flush(c2, segs, m0 + 32 + hi * 8, aggr, f, o_sum, o_min, o_max, o_mean, o_m1, o_m2);
        aggr_flush(c3, segs, m0 + 48 + hi * 8, aggr, f, o_sum, o_min, o_max, o_mean, o_m1, o_m2);
    }
}

// ---- kernel 4: finalize z + GEMM2, 8 graphs per block (8x W2 reuse) ----
__launch_bounds__(256)
__global__ void k_finalize(const float* __restrict__ o_sum, const unsigned* __restrict__ o_min,
                           const unsigned* __restrict__ o_max, const float* __restrict__ o_mean,
                           const float* __restrict__ o_m1, const float* __restrict__ o_m2,
                           const int* __restrict__ cnt, const float* __restrict__ W2,
                           const float* __restrict__ bias, float* __restrict__ out) {
    __shared__ float zs[GPB][AK];
    const int g0 = blockIdx.x * GPB;

    for (int idx = threadIdx.x; idx < GPB * AK; idx += 256) {
        const int g = idx / AK;
        const int j = idx - g * AK;
        const int b = g0 + g;
        const float cn = fmaxf((float)cnt[b], 1.0f);
        const int ag = j >> 6, f = j & 63;
        const int o = b * AC + f;
        float zv;
        if      (ag == 0) zv = o_sum[o];
        else if (ag == 1) zv = decf(o_min[o]);
        else if (ag == 2) zv = decf(o_max[o]);
        else if (ag == 3) zv = o_mean[o] / cn;
        else {
            const float m  = o_m1[o] / cn;
            const float m2 = o_m2[o] / cn;
            zv = sqrtf(fmaxf(m2 - m * m, 1e-5f));
        }
        zs[g][j] = zv;
    }
    __syncthreads();

    const int t = threadIdx.x;            // output column
    float acc[GPB];
#pragma unroll
    for (int g = 0; g < GPB; ++g) acc[g] = bias[t];
    for (int j = 0; j < AK; ++j) {
        const float w = W2[j * HC + t];   // loaded once, reused for 8 graphs
#pragma unroll
        for (int g = 0; g < GPB; ++g) acc[g] = fmaf(zs[g][j], w, acc[g]);
    }
#pragma unroll
    for (int g = 0; g < GPB; ++g) out[(g0 + g) * HC + t] = acc[g];
}

extern "C" void kernel_launch(void* const* d_in, const int* in_sizes, int n_in,
                              void* d_out, int out_size, void* d_ws, size_t ws_size,
                              hipStream_t stream) {
    const float* x     = (const float*)d_in[0];
    const int*   batch = (const int*)  d_in[1];
    // d_in[2] = num_graphs scalar (compile-time NSEG)
    const float* W1    = (const float*)d_in[3];
    const float* W2    = (const float*)d_in[4];
    const float* bias  = (const float*)d_in[5];
    float* out = (float*)d_out;

    char* ws = (char*)d_ws;
    size_t off = 0;
    auto carve = [&](size_t bytes) -> void* {
        off = (off + 255) & ~(size_t)255;
        void* p = ws + off;
        off += bytes;
        return p;
    };
    _Float16* Wt    = (_Float16*)carve((size_t)AK * HC * sizeof(_Float16));
    float*    o_sum = (float*)   carve((size_t)NSEG * AC * sizeof(float));
    unsigned* o_min = (unsigned*)carve((size_t)NSEG * AC * sizeof(unsigned));
    unsigned* o_max = (unsigned*)carve((size_t)NSEG * AC * sizeof(unsigned));
    float*    o_mean= (float*)   carve((size_t)NSEG * AC * sizeof(float));
    float*    o_m1  = (float*)   carve((size_t)NSEG * AC * sizeof(float));
    float*    o_m2  = (float*)   carve((size_t)NSEG * AC * sizeof(float));
    int*      cnt   = (int*)     carve((size_t)NSEG * sizeof(int));

    k_transpose_w1<<<AK, HC, 0, stream>>>(W1, Wt);
    k_init<<<(NSEG * AC + 255) / 256, 256, 0, stream>>>(o_sum, o_min, o_max,
                                                        o_mean, o_m1, o_m2, cnt);
    k_gemm_aggr<<<NBLK, 256, 0, stream>>>(x, batch, Wt, o_sum, o_min, o_max,
                                          o_mean, o_m1, o_m2, cnt);
    k_finalize<<<NSEG / GPB, 256, 0, stream>>>(o_sum, o_min, o_max, o_mean, o_m1, o_m2,
                                               cnt, W2, bias, out);
}
